// TAGConv_3l_128h_wnw_k3_52896817218189
// MI455X (gfx1250) — matrix operand order repaired and checked by reading, not compiled, not run on hardware
//
#include <hip/hip_runtime.h>
#include <hip/hip_bf16.h>
#include <math.h>

#define NN 50000
#define NE 600000
#define HID 128
#define NCLS 26

typedef __attribute__((ext_vector_type(16))) __bf16 v16bf;
typedef __attribute__((ext_vector_type(8)))  float  v8f;
typedef __attribute__((ext_vector_type(8)))  unsigned int v8u;

// Native conversion: compiler lowers to hardware bf16 cvt (RNE) on gfx1250.
__device__ __forceinline__ __bf16 cvt_bf16(float f) { return (__bf16)f; }

// ---------------------------------------------------------------- utility
__global__ void fill0_kernel(float* __restrict__ p, long long n) {
    long long i = (long long)blockIdx.x * blockDim.x + threadIdx.x;
    long long stride = (long long)gridDim.x * blockDim.x;
    for (; i < n; i += stride) p[i] = 0.0f;
}

// ---------------------------------------------------------------- degrees
__global__ void deg_kernel(const int* __restrict__ dst, const float* __restrict__ ew,
                           float* __restrict__ degw, float* __restrict__ degnw, int E) {
    int e = blockIdx.x * blockDim.x + threadIdx.x;
    if (e >= E) return;
    int d = dst[e];
    atomicAdd(&degw[d], ew[e]);
    atomicAdd(&degnw[d], 1.0f);
}

__global__ void dinv_kernel(float* __restrict__ degw, float* __restrict__ degnw, int n) {
    int i = blockIdx.x * blockDim.x + threadIdx.x;
    if (i >= n) return;
    float a = degw[i];
    float b = degnw[i];
    degw[i]  = a > 0.0f ? rsqrtf(a) : 0.0f;
    degnw[i] = b > 0.0f ? rsqrtf(b) : 0.0f;
}

__global__ void norm_kernel(const int* __restrict__ src, const int* __restrict__ dst,
                            const float* __restrict__ ew,
                            const float* __restrict__ dinvw, const float* __restrict__ dinvnw,
                            float* __restrict__ normw, float* __restrict__ normnw, int E) {
    int e = blockIdx.x * blockDim.x + threadIdx.x;
    if (e >= E) return;
    int s = src[e], d = dst[e];
    normw[e]  = dinvw[s] * ew[e] * dinvw[d];
    normnw[e] = dinvnw[s] * dinvnw[d];
}

// ---------------------------------------------------------------- SpMM: out[dst] += norm * h[src]
// 32 consecutive lanes = 1 edge (norm/src/dst loads are wave-broadcast);
// each lane handles a float4 of the 128-wide feature row.
__global__ void spmm_kernel(const float* __restrict__ h,
                            const int* __restrict__ src, const int* __restrict__ dst,
                            const float* __restrict__ norm,
                            float* __restrict__ out, int E) {
    long long t = (long long)blockIdx.x * blockDim.x + threadIdx.x;
    int e = (int)(t >> 5);
    int q = (int)(t & 31);
    if (e >= E) return;
    float w = norm[e];
    int s = src[e], d = dst[e];
    float4 v = *(const float4*)(h + (size_t)s * HID + q * 4);
    float* o = out + (size_t)d * HID + q * 4;
    atomicAdd(o + 0, w * v.x);
    atomicAdd(o + 1, w * v.y);
    atomicAdd(o + 2, w * v.z);
    atomicAdd(o + 3, w * v.w);
}

// ---------------------------------------------------------------- WMMA GEMM: out (+)= h @ W  [nrows x 128 @ 128 x 128]
// 256 threads = 8 waves; each wave owns a 16-row strip x 128 cols (8 v8f acc tiles).
// mode 0: out = h@W   mode 1: out += h@W   mode 2: act = elu(out + h@W + bias)
__global__ __launch_bounds__(256) void gemm_tag_kernel(
    const float* __restrict__ h, const float* __restrict__ W,
    float* __restrict__ out, const float* __restrict__ bias,
    float* __restrict__ act, int nrows, int mode)
{
    // W staged in LDS as bf16 in exact B-fragment lane order:
    // idx = ((kt*8+ct)*32 + lane)*16 + e  with  K = kt*32 + 16*(lane/16) + e, Ncol = ct*16 + lane%16
    __shared__ unsigned short sW[16384];  // 128x128 bf16 = 32 KB

    int t = threadIdx.x;
    for (int widx = t; widx < 16384; widx += 256) {       // coalesced global read of W
        int K  = widx >> 7;
        int nn = widx & 127;
        int kt = K >> 5, kr = K & 31;
        int g  = kr >> 4, e = kr & 15;
        int ct = nn >> 4, nl = nn & 15;
        int lds = (((kt * 8 + ct) * 32) + (g * 16 + nl)) * 16 + e;
        sW[lds] = __builtin_bit_cast(unsigned short, cvt_bf16(W[widx]));
    }
    __syncthreads();

    int wave = t >> 5;
    int lane = t & 31;
    int r0 = (blockIdx.x * 8 + wave) * 16;
    if (r0 >= nrows) return;

    int g = lane >> 4;      // half-wave
    int m = lane & 15;

    v8f acc[8];
    #pragma unroll
    for (int ct = 0; ct < 8; ++ct) {
        #pragma unroll
        for (int r = 0; r < 8; ++r) acc[ct][r] = 0.0f;
    }

    const float* hrow = h + (size_t)(r0 + m) * HID;   // A row for this lane

    #pragma unroll
    for (int kt = 0; kt < 4; ++kt) {
        // A fragment (16x32 bf16):  e0-15 -> K = kt*32+16g+e
        const float4* hp = (const float4*)(hrow + kt * 32);
        float4 q0 = hp[4 * g];
        float4 q1 = hp[4 * g + 1];
        float4 q2 = hp[4 * g + 2];
        float4 q3 = hp[4 * g + 3];
        v16bf afrag;
        afrag[0]  = cvt_bf16(q0.x); afrag[1]  = cvt_bf16(q0.y);
        afrag[2]  = cvt_bf16(q0.z); afrag[3]  = cvt_bf16(q0.w);
        afrag[4]  = cvt_bf16(q1.x); afrag[5]  = cvt_bf16(q1.y);
        afrag[6]  = cvt_bf16(q1.z); afrag[7]  = cvt_bf16(q1.w);
        afrag[8]  = cvt_bf16(q2.x); afrag[9]  = cvt_bf16(q2.y);
        afrag[10] = cvt_bf16(q2.z); afrag[11] = cvt_bf16(q2.w);
        afrag[12] = cvt_bf16(q3.x); afrag[13] = cvt_bf16(q3.y);
        afrag[14] = cvt_bf16(q3.z); afrag[15] = cvt_bf16(q3.w);

        #pragma unroll
        for (int ct = 0; ct < 8; ++ct) {
            v8u raw = *(const v8u*)&sW[(((kt * 8 + ct) * 32) + lane) * 16];
            v16bf bfrag = __builtin_bit_cast(v16bf, raw);
            acc[ct] = __builtin_amdgcn_wmma_f32_16x16x32_bf16(
                false, afrag, false, bfrag, (short)0, acc[ct], false, false);
        }
    }

    // Epilogue: C/D layout -> M = 8g + r, N = ct*16 + m
    #pragma unroll
    for (int ct = 0; ct < 8; ++ct) {
        int ncol = ct * 16 + m;
        #pragma unroll
        for (int r = 0; r < 8; ++r) {
            int orow = r0 + g * 8 + r;
            size_t off = (size_t)orow * HID + ncol;
            float v = acc[ct][r];
            if (mode == 0) {
                out[off] = v;
            } else if (mode == 1) {
                out[off] += v;
            } else {
                float s = out[off] + v + bias[ncol];
                act[off] = s > 0.0f ? s : (expf(s) - 1.0f);
            }
        }
    }
}

// ---------------------------------------------------------------- classifier: out = [xw|xnw] @ clfW + clfb
__global__ void clf_kernel(const float* __restrict__ xw, const float* __restrict__ xnw,
                           const float* __restrict__ W, const float* __restrict__ b,
                           float* __restrict__ out, int n) {
    int node = blockIdx.x;
    int c = threadIdx.x;              // blockDim = 32; lanes 26..31 idle
    if (node >= n || c >= NCLS) return;
    const float* r1 = xw  + (size_t)node * HID;
    const float* r2 = xnw + (size_t)node * HID;
    float s = b[c];
    #pragma unroll 4
    for (int k = 0; k < HID; ++k) s += r1[k] * W[k * NCLS + c];
    #pragma unroll 4
    for (int k = 0; k < HID; ++k) s += r2[k] * W[(HID + k) * NCLS + c];
    out[(size_t)node * NCLS + c] = s;
}

// ---------------------------------------------------------------- launcher
extern "C" void kernel_launch(void* const* d_in, const int* in_sizes, int n_in,
                              void* d_out, int out_size, void* d_ws, size_t ws_size,
                              hipStream_t stream) {
    const float* x    = (const float*)d_in[0];
    const int*   src  = (const int*)d_in[1];
    const int*   dst  = (const int*)d_in[2];
    const float* ew   = (const float*)d_in[3];
    const float* Ww   = (const float*)d_in[4];
    const float* bw   = (const float*)d_in[5];
    const float* Wnw  = (const float*)d_in[6];
    const float* bnw  = (const float*)d_in[7];
    const float* clfW = (const float*)d_in[8];
    const float* clfb = (const float*)d_in[9];
    float* outp = (float*)d_out;

    const int N = NN, E = NE;
    float* ws     = (float*)d_ws;
    float* degw   = ws;                       // N  (becomes dinv_w)
    float* degnw  = degw + N;                 // N  (becomes dinv_nw)
    float* normw  = degnw + N;                // E
    float* normnw = normw + E;                // E
    float* hA     = normnw + E;               // N*128
    float* hB     = hA + (size_t)N * HID;     // N*128
    float* outbuf = hB + (size_t)N * HID;     // N*128
    float* xwb    = outbuf + (size_t)N * HID; // N*128
    float* xnwb   = xwb + (size_t)N * HID;    // N*128

    // normalization precompute
    fill0_kernel<<<256, 256, 0, stream>>>(degw, (long long)2 * N);
    deg_kernel<<<(E + 255) / 256, 256, 0, stream>>>(dst, ew, degw, degnw, E);
    dinv_kernel<<<(N + 255) / 256, 256, 0, stream>>>(degw, degnw, N);
    norm_kernel<<<(E + 255) / 256, 256, 0, stream>>>(src, dst, ew, degw, degnw, normw, normnw, E);

    const int gemm_grid = (N + 127) / 128;    // 256-thread blocks, 8 waves, 128 rows
    long long spmm_threads = (long long)E * 32;
    int spmm_blocks = (int)((spmm_threads + 255) / 256);

    for (int br = 0; br < 2; ++br) {
        const float* Wset = (br == 0) ? Ww  : Wnw;
        const float* bset = (br == 0) ? bw  : bnw;
        const float* nrm  = (br == 0) ? normw : normnw;
        float* xbuf       = (br == 0) ? xwb : xnwb;

        const float* hin = x;
        for (int l = 0; l < 3; ++l) {
            const float* Wl = Wset + (size_t)l * 4 * HID * HID;
            const float* bl = bset + (size_t)l * HID;

            // hop 0: out = hin @ W0
            gemm_tag_kernel<<<gemm_grid, 256, 0, stream>>>(hin, Wl, outbuf,
                                                           nullptr, nullptr, N, 0);
            const float* hsrc = hin;
            float* ping = hA;
            float* pong = hB;
            for (int k = 1; k <= 3; ++k) {
                fill0_kernel<<<2048, 256, 0, stream>>>(ping, (long long)N * HID);
                spmm_kernel<<<spmm_blocks, 256, 0, stream>>>(hsrc, src, dst, nrm, ping, E);
                gemm_tag_kernel<<<gemm_grid, 256, 0, stream>>>(
                    ping, Wl + (size_t)k * HID * HID, outbuf, bl, xbuf, N,
                    (k < 3) ? 1 : 2);
                hsrc = ping;
                float* tp = ping; ping = pong; pong = tp;
            }
            hin = xbuf;
        }
    }

    clf_kernel<<<N, 32, 0, stream>>>(xwb, xnwb, clfW, clfb, outp, N);
}
